// EdgePassing_22625887715781
// MI455X (gfx1250) — matrix operand order repaired and checked by reading, not compiled, not run on hardware
//
#include <hip/hip_runtime.h>

typedef __attribute__((ext_vector_type(16))) _Float16 v16h;
typedef __attribute__((ext_vector_type(8)))  _Float16 v8h;
typedef __attribute__((ext_vector_type(8)))  float    v8f;

#define DIM 64

// ---- prep kernel 1: f32 -> f16 (features table; stays L2-resident) ----
__global__ void cvt_f32_to_f16(const float* __restrict__ src,
                               _Float16* __restrict__ dst, int n) {
    int i = blockIdx.x * blockDim.x + threadIdx.x;
    int stride = gridDim.x * blockDim.x;
    for (; i < n; i += stride) dst[i] = (_Float16)src[i];
}

// ---- prep kernel 2: W [k][n] f32 -> Wt [n][k] f16 (B-operand friendly) ----
__global__ void transpose_w_f16(const float* __restrict__ W,
                                _Float16* __restrict__ Wt) {
    for (int t = threadIdx.x; t < DIM * DIM; t += blockDim.x) {
        int n = t >> 6, k = t & 63;
        Wt[n * DIM + k] = (_Float16)W[k * DIM + n];
    }
}

// ---- main kernel: gather + 2x (GEMM + bias + ReLU), weights register-resident ----
__global__ __launch_bounds__(256)
void edge_mlp_wmma(const _Float16* __restrict__ feat16,
                   const long long* __restrict__ src_idx,
                   const _Float16* __restrict__ W1t,
                   const float*    __restrict__ b1,
                   const _Float16* __restrict__ W2t,
                   const float*    __restrict__ b2,
                   float* __restrict__ out,
                   int E) {
    // LDS only for the per-wave layer-1 output (layout shuffle D->A)
    __shared__ __align__(32) _Float16 sH[8][16 * DIM];

    const int lane = threadIdx.x & 31;
    const int wave = threadIdx.x >> 5;
    const int l    = lane & 15;   // row (edge) / column (N) within tile
    const int hi   = lane >> 4;   // lane-half selector
    _Float16* myH = sH[wave];

    // ---- hoist all weight B-operands into registers (loop-invariant) ----
    // B operand (contiguous k order, same as A): lane l holds column N=l; low lane-half K 0-15,
    // high half K 16-31 -> one contiguous 32B read of W^T row per operand.
    v16h w1b[4][2], w2b[4][2];
    float bias1[4], bias2[4];
#pragma unroll
    for (int nt = 0; nt < 4; ++nt) {
        const _Float16* r1 = W1t + (nt * 16 + l) * DIM + hi * 16;
        const _Float16* r2 = W2t + (nt * 16 + l) * DIM + hi * 16;
#pragma unroll
        for (int ks = 0; ks < 2; ++ks) {
            w1b[nt][ks] = *(const v16h*)(r1 + ks * 32);
            w2b[nt][ks] = *(const v16h*)(r2 + ks * 32);
        }
        bias1[nt] = b1[nt * 16 + l];
        bias2[nt] = b2[nt * 16 + l];
    }

    const int numChunks = (E + 127) >> 7;   // 128 edges per block-iteration
    for (int c = blockIdx.x; c < numChunks; c += gridDim.x) {
        const int e0 = (c << 7) + wave * 16;   // wave-uniform
        if (e0 >= E) continue;

        // ---- gather: directly into WMMA A layout (16-bit A 16x32 striping) ----
        int eRow = e0 + l;
        if (eRow >= E) eRow = E - 1;           // clamp; store is guarded below
        const _Float16* fp = feat16 + (long long)src_idx[eRow] * DIM;

        union { v16h v; v8h h[2]; } a0, a1;
        a0.h[0] = *(const v8h*)(fp +      hi * 16);   // K 0-7  / 16-23
        a0.h[1] = *(const v8h*)(fp +  8 + hi * 16);   // K 8-15 / 24-31
        a1.h[0] = *(const v8h*)(fp + 32 + hi * 16);   // K 32-39 / 48-55
        a1.h[1] = *(const v8h*)(fp + 40 + hi * 16);   // K 40-47 / 56-63

        // ---- layer 1: h = relu(x @ W1 + b1); 8 back-to-back WMMAs ----
        v8f acc[4];
#pragma unroll
        for (int nt = 0; nt < 4; ++nt) {
            v8f cz = {};
            cz = __builtin_amdgcn_wmma_f32_16x16x32_f16(false, a0.v, false,
                     w1b[nt][0], (short)0, cz, false, false);
            cz = __builtin_amdgcn_wmma_f32_16x16x32_f16(false, a1.v, false,
                     w1b[nt][1], (short)0, cz, false, false);
            acc[nt] = cz;
        }

        // bias + relu, spill h to per-wave LDS slab (row-major 16x64 f16)
#pragma unroll
        for (int nt = 0; nt < 4; ++nt) {
            float bias = bias1[nt];
#pragma unroll
            for (int v = 0; v < 8; ++v) {
                float x = fmaxf(acc[nt][v] + bias, 0.0f);
                myH[(v + 8 * hi) * DIM + nt * 16 + l] = (_Float16)x;
            }
        }
        // LDS ops are in-order within a wave: safe to read h back now.

        // ---- layer 2 A operands from LDS (same A striping) ----
        union { v16h v; v8h h[2]; } c0, c1;
        const _Float16* hp = myH + l * DIM;
        c0.h[0] = *(const v8h*)(hp +      hi * 16);
        c0.h[1] = *(const v8h*)(hp +  8 + hi * 16);
        c1.h[0] = *(const v8h*)(hp + 32 + hi * 16);
        c1.h[1] = *(const v8h*)(hp + 40 + hi * 16);

        // ---- layer 2: out = relu(h @ W2 + b2), store f32 ----
#pragma unroll
        for (int nt = 0; nt < 4; ++nt) {
            v8f cz = {};
            cz = __builtin_amdgcn_wmma_f32_16x16x32_f16(false, c0.v, false,
                     w2b[nt][0], (short)0, cz, false, false);
            cz = __builtin_amdgcn_wmma_f32_16x16x32_f16(false, c1.v, false,
                     w2b[nt][1], (short)0, cz, false, false);
            float bias = bias2[nt];
#pragma unroll
            for (int v = 0; v < 8; ++v) {
                int m = e0 + v + 8 * hi;           // edge row (C/D layout)
                if (m < E) {
                    out[(long long)m * DIM + nt * 16 + l] =
                        fmaxf(cz[v] + bias, 0.0f);
                }
            }
        }
    }
}

extern "C" void kernel_launch(void* const* d_in, const int* in_sizes, int n_in,
                              void* d_out, int out_size, void* d_ws, size_t ws_size,
                              hipStream_t stream) {
    const float*     features = (const float*)d_in[0];
    const long long* src_idx  = (const long long*)d_in[1];   // jnp.int64
    const float*     W1       = (const float*)d_in[2];
    const float*     b1       = (const float*)d_in[3];
    const float*     W2       = (const float*)d_in[4];
    const float*     b2       = (const float*)d_in[5];
    float* out = (float*)d_out;

    const int featElems = in_sizes[0];   // N * 64
    const int E         = in_sizes[1];   // edge count

    // workspace layout: [features f16][W1^T f16][W2^T f16]
    _Float16* feat16 = (_Float16*)d_ws;
    _Float16* W1t    = feat16 + featElems;
    _Float16* W2t    = W1t + DIM * DIM;

    cvt_f32_to_f16<<<256, 256, 0, stream>>>(features, feat16, featElems);
    transpose_w_f16<<<1, 256, 0, stream>>>(W1, W1t);
    transpose_w_f16<<<1, 256, 0, stream>>>(W2, W2t);

    const int numChunks = (E + 127) / 128;
    const int grid = numChunks < 1024 ? numChunks : 1024;
    edge_mlp_wmma<<<grid, 256, 0, stream>>>(feat16, src_idx, W1t, b1, W2t, b2,
                                            out, E);
}
